// LSTM_23922967838797
// MI455X (gfx1250) — matrix operand order repaired and checked by reading, not compiled, not run on hardware
//
#include <hip/hip_runtime.h>
#include <hip/hip_bf16.h>
#include <stdint.h>

// ---------------------------------------------------------------------------
// LSTM cell step, B=2048 H=1024, bf16 WMMA GEMM with fused gate epilogue.
// gates = [input|hidden] @ [[Wf Wi Wo Wc],[Uf Ui Uo Uc]] + b ; LSTM update.
// BM=256 x BN=64 tile (x4 gates), KT=64 LDS stages. Each of 16 waves owns a
// 32x32 output patch (2 m-subtiles x 2 n-subtiles x 4 gates) so every LDS
// B-fragment feeds two WMMAs and every A-fragment feeds eight. Register
// double-buffered global->LDS staging + global_prefetch for the K+2 stage.
// ---------------------------------------------------------------------------

typedef __attribute__((ext_vector_type(16))) __bf16 v16bf;
typedef __attribute__((ext_vector_type(8)))  float  v8f;

#define B_DIM   2048
#define H_DIM   1024
#define K_TOT   (2 * H_DIM)
#define BM      256          // batch rows per workgroup
#define BN      64           // h columns per workgroup (x4 gates internally)
#define KT      64           // K tile staged in LDS
#define PAD_DW  36           // padded LDS row: 72 bf16 = 36 dwords (conflict-free)
#define NTHREADS 512         // 16 waves

struct U8 { uint4 lo, hi; };

static __device__ __forceinline__ uint32_t f32_to_bf16_rne(float f) {
  uint32_t u = __builtin_bit_cast(uint32_t, f);
  return (u + 0x7FFFu + ((u >> 16) & 1u)) >> 16;
}
static __device__ __forceinline__ uint32_t pack_bf16(float a, float b) {
  return f32_to_bf16_rne(a) | (f32_to_bf16_rne(b) << 16);
}

// A fragment (16x32 bf16): lanes 0-15 -> M=lane, K = kk+[0..7] & kk+[16..23]
//                          lanes 16-31 -> M=lane-16, K = kk+[8..15] & kk+[24..31]
static __device__ __forceinline__ v16bf load_frag_a(const uint32_t* base, int row,
                                                    int kk, int lane) {
  int d0 = row * PAD_DW + (kk >> 1) + ((lane >= 16) ? 4 : 0);
  const uint4* p = (const uint4*)(base + d0);
  U8 u{p[0], p[2]};                       // dwords [d0..d0+3], [d0+8..d0+11]
  return __builtin_bit_cast(v16bf, u);
}

// B fragment (32x16 bf16, stored column-major in LDS as [n][k]):
//   lanes 0-15  -> N=lane,    K = kk+[0..7] & kk+[16..23]
//   lanes 16-31 -> N=lane-16, K = kk+[8..15] & kk+[24..31]
static __device__ __forceinline__ v16bf load_frag_b(const uint32_t* base, int row,
                                                    int kk, int lane) {
  int d0 = row * PAD_DW + (kk >> 1) + ((lane >= 16) ? 4 : 0);
  const uint4* p = (const uint4*)(base + d0);
  U8 u{p[0], p[2]};                       // dwords [d0..d0+3], [d0+8..d0+11]
  return __builtin_bit_cast(v16bf, u);
}

static __device__ __forceinline__ float sigm(float x) {
  return 1.0f / (1.0f + __expf(-x));
}
static __device__ __forceinline__ float tanh_fast(float x) {
  return 1.0f - 2.0f / (__expf(2.0f * x) + 1.0f);
}

__global__ __launch_bounds__(NTHREADS)
void lstm_wmma_kernel(const float* __restrict__ input,
                      const float* __restrict__ hidden,
                      const float* __restrict__ pre_cell,
                      const float* __restrict__ w_i, const float* __restrict__ w_f,
                      const float* __restrict__ w_o, const float* __restrict__ w_c,
                      const float* __restrict__ u_i, const float* __restrict__ u_f,
                      const float* __restrict__ u_o, const float* __restrict__ u_c,
                      const float* __restrict__ bias1, const float* __restrict__ bias2,
                      const float* __restrict__ bias3, const float* __restrict__ bias4,
                      const float* __restrict__ drop_mask,
                      float* __restrict__ out) {
  __shared__ uint32_t As[BM * PAD_DW];        // 36 KB : A tile, bf16 pairs, row-major
  __shared__ uint32_t Bs[4][BN * PAD_DW];     // 36 KB : per-gate B tile, [n][k] bf16

  const int tid  = threadIdx.x;
  const int lane = tid & 31;
  const int wave = tid >> 5;                  // 0..15
  const int m0   = blockIdx.x * BM;
  const int n0   = blockIdx.y * BN;

  // wave -> 32x32 output patch: rows [wrow*32, +32), cols [wcol*32, +32)
  const int wrow = wave & 7;                  // 8 row groups of 32 batch rows
  const int wcol = wave >> 3;                 // 2 col groups of 32 h columns
  const int frow = lane & 15;                 // fragment row within 16

  // gate order matches reference concat: f, i, o, c  with biases 1..4
  const float* wxs[4] = {w_f, w_i, w_o, w_c};
  const float* wus[4] = {u_f, u_i, u_o, u_c};

  // acc[gate][nt][mt] : nt,mt in {0,1} -> 16 x v8f = 128 VGPRs
  v8f acc[4][2][2];
#pragma unroll
  for (int g = 0; g < 4; ++g)
#pragma unroll
    for (int nt = 0; nt < 2; ++nt)
#pragma unroll
      for (int mt = 0; mt < 2; ++mt)
        acc[g][nt][mt] = (v8f)0.0f;

  // -------- staging index precompute --------
  const int a_kp  = lane;                     // A: (row = wave + it*16, kpair = lane)
  const int b_n   = tid & 63;                 // B: (n = tid&63, kpair = (tid>>6)+i*8)
  const int b_kp0 = tid >> 6;                 // 0..7

  float2 aReg[16];
  float  bLo[4][4], bHi[4][4];

  // -------- prologue: prefetch stage 0 into registers --------
  {
#pragma unroll
    for (int it = 0; it < 16; ++it) {
      const int row = wave + it * 16;
      aReg[it] = *(const float2*)(input + (size_t)(m0 + row) * H_DIM + 2 * a_kp);
    }
#pragma unroll
    for (int g = 0; g < 4; ++g) {
      const float* wg = wxs[g];
#pragma unroll
      for (int i = 0; i < 4; ++i) {
        const int k = 2 * (b_kp0 + i * 8);
        bLo[g][i] = wg[(size_t)k * H_DIM + n0 + b_n];
        bHi[g][i] = wg[(size_t)(k + 1) * H_DIM + n0 + b_n];
      }
    }
  }

  for (int k0 = 0; k0 < K_TOT; k0 += KT) {
    // ---- commit prefetched stage to LDS (fp32 -> packed bf16) ----
#pragma unroll
    for (int it = 0; it < 16; ++it) {
      const int row = wave + it * 16;
      As[row * PAD_DW + a_kp] = pack_bf16(aReg[it].x, aReg[it].y);
    }
#pragma unroll
    for (int g = 0; g < 4; ++g)
#pragma unroll
      for (int i = 0; i < 4; ++i)
        Bs[g][b_n * PAD_DW + (b_kp0 + i * 8)] = pack_bf16(bLo[g][i], bHi[g][i]);
    __syncthreads();

    // ---- issue next stage's global loads (overlaps with WMMA below) ----
    const int k1 = k0 + KT;
    if (k1 < K_TOT) {
      const float* Asrc = (k1 < H_DIM) ? input : hidden;
      const float* const* W = (k1 < H_DIM) ? wxs : wus;
      const int kb = k1 & (H_DIM - 1);
#pragma unroll
      for (int it = 0; it < 16; ++it) {
        const int row = wave + it * 16;
        aReg[it] = *(const float2*)(Asrc + (size_t)(m0 + row) * H_DIM + kb + 2 * a_kp);
      }
#pragma unroll
      for (int g = 0; g < 4; ++g) {
        const float* wg = W[g];
#pragma unroll
        for (int i = 0; i < 4; ++i) {
          const int k = kb + 2 * (b_kp0 + i * 8);
          bLo[g][i] = wg[(size_t)k * H_DIM + n0 + b_n];
          bHi[g][i] = wg[(size_t)(k + 1) * H_DIM + n0 + b_n];
        }
      }
    }
    // ---- L2 warm-up for stage k0+2 (global_prefetch_b8) ----
    const int k2 = k0 + 2 * KT;
    if (k2 < K_TOT) {
      const float* Asrc2 = (k2 < H_DIM) ? input : hidden;
      const float* const* W2 = (k2 < H_DIM) ? wxs : wus;
      const int kb2 = k2 & (H_DIM - 1);
      __builtin_prefetch(Asrc2 + (size_t)(m0 + wave) * H_DIM + kb2 + 2 * a_kp, 0, 1);
#pragma unroll
      for (int g = 0; g < 4; ++g)
        __builtin_prefetch(W2[g] + (size_t)(kb2 + 2 * b_kp0) * H_DIM + n0 + b_n, 0, 1);
    }

    // ---- compute: per kk-step, 2 A-frags + 8 B-frags -> 16 WMMAs.
    //      Each B fragment feeds two WMMAs (mt=0,1) -> only one dscnt wait
    //      per WMMA pair; each A fragment feeds eight WMMAs.
#pragma unroll
    for (int kk = 0; kk < KT; kk += 32) {
      const v16bf a0 = load_frag_a(As, wrow * 32 + 0 * 16 + frow, kk, lane);
      const v16bf a1 = load_frag_a(As, wrow * 32 + 1 * 16 + frow, kk, lane);
#pragma unroll
      for (int g = 0; g < 4; ++g) {
#pragma unroll
        for (int nt = 0; nt < 2; ++nt) {
          const v16bf bf =
              load_frag_b(Bs[g], wcol * 32 + nt * 16 + frow, kk, lane);
          acc[g][nt][0] = __builtin_amdgcn_wmma_f32_16x16x32_bf16(
              false, a0, false, bf, (short)0, acc[g][nt][0], false, false);
          acc[g][nt][1] = __builtin_amdgcn_wmma_f32_16x16x32_bf16(
              false, a1, false, bf, (short)0, acc[g][nt][1], false, false);
        }
      }
    }
    __syncthreads();
  }

  // ---- fused LSTM epilogue. C/D layout: VGPR r, lanes 0-15 -> M=r, N=lane;
  //      lanes 16-31 -> M=r+8, N=lane-16.
  float* out_h = out;
  float* out_c = out + (size_t)B_DIM * H_DIM;
#pragma unroll
  for (int nt = 0; nt < 2; ++nt) {
    const int col = n0 + wcol * 32 + nt * 16 + frow;
    const float b1 = bias1[col];
    const float b2 = bias2[col];
    const float b3 = bias3[col];
    const float b4 = bias4[col];
#pragma unroll
    for (int mt = 0; mt < 2; ++mt) {
#pragma unroll
      for (int r = 0; r < 8; ++r) {
        const int brow = m0 + wrow * 32 + mt * 16 + r + ((lane >= 16) ? 8 : 0);
        const size_t idx = (size_t)brow * H_DIM + col;
        const float f_t = sigm(acc[0][nt][mt][r] + b1);
        const float i_t = sigm(acc[1][nt][mt][r] + b2);
        const float o_t = sigm(acc[2][nt][mt][r] + b3);
        const float g_t = tanh_fast(acc[3][nt][mt][r] + b4);
        const float c_t = pre_cell[idx] * f_t + i_t * g_t;
        const float h_t = o_t * tanh_fast(c_t) * drop_mask[idx];
        out_h[idx] = h_t;
        out_c[idx] = c_t;
      }
    }
  }
}

extern "C" void kernel_launch(void* const* d_in, const int* in_sizes, int n_in,
                              void* d_out, int out_size, void* d_ws, size_t ws_size,
                              hipStream_t stream) {
  (void)in_sizes; (void)n_in; (void)out_size; (void)d_ws; (void)ws_size;
  const float* input     = (const float*)d_in[0];
  const float* hidden    = (const float*)d_in[1];
  const float* pre_cell  = (const float*)d_in[2];
  const float* w_i       = (const float*)d_in[3];
  const float* w_f       = (const float*)d_in[4];
  const float* w_o       = (const float*)d_in[5];
  const float* w_c       = (const float*)d_in[6];
  const float* u_i       = (const float*)d_in[7];
  const float* u_f       = (const float*)d_in[8];
  const float* u_o       = (const float*)d_in[9];
  const float* u_c       = (const float*)d_in[10];
  const float* bias1     = (const float*)d_in[11];
  const float* bias2     = (const float*)d_in[12];
  const float* bias3     = (const float*)d_in[13];
  const float* bias4     = (const float*)d_in[14];
  const float* drop_mask = (const float*)d_in[15];

  dim3 grid(B_DIM / BM, H_DIM / BN);   // (8, 16)
  dim3 block(NTHREADS);
  lstm_wmma_kernel<<<grid, block, 0, stream>>>(
      input, hidden, pre_cell, w_i, w_f, w_o, w_c, u_i, u_f, u_o, u_c,
      bias1, bias2, bias3, bias4, drop_mask, (float*)d_out);
}
